// GAT_layer2_3255585210652
// MI455X (gfx1250) — matrix operand order repaired and checked by reading, not compiled, not run on hardware
//
#include <hip/hip_runtime.h>
#include <hip/hip_bf16.h>
#include <math.h>

// ---------------------------------------------------------------------------
// 2-layer GAT for MI455X (gfx1250, wave32).
// Dense h@W GEMMs use v_wmma_f32_16x16x32_f16 (K=32 == one WMMA).
// Edge softmax/aggregation uses the standard 3-pass atomic scheme.
// ---------------------------------------------------------------------------

typedef _Float16 v16h __attribute__((ext_vector_type(16)));
typedef float    v8f  __attribute__((ext_vector_type(8)));
typedef float    v4f  __attribute__((ext_vector_type(4)));

#define NEG_SLOPE 0.2f

static inline int ceil_div(long long a, long long b) { return (int)((a + b - 1) / b); }

// ---------------- fill (init) ----------------
__global__ void gat_fill_u32(unsigned int* __restrict__ p, unsigned int v, long long n) {
    long long i = (long long)blockIdx.x * blockDim.x + threadIdx.x;
    if (i < n) p[i] = v;
}

// ---------------- argmax over 128 features + embedding gather ----------------
// one wave (32 lanes) per node
__global__ void gat_embed(const float* __restrict__ x, const float* __restrict__ emb,
                          float* __restrict__ h0, int N) {
    int wid  = (int)((blockIdx.x * (long long)blockDim.x + threadIdx.x) >> 5);
    int lane = threadIdx.x & 31;
    if (wid >= N) return;
    const float* xr = x + (size_t)wid * 128;
    float best = -3.4e38f; int bidx = 0;
    #pragma unroll
    for (int j = 0; j < 4; ++j) {
        int idx = lane + 32 * j;           // ascending -> strict > keeps first occurrence
        float v = xr[idx];
        if (v > best) { best = v; bidx = idx; }
    }
    #pragma unroll
    for (int off = 16; off > 0; off >>= 1) {
        float ov = __shfl_xor(best, off, 32);
        int   oi = __shfl_xor(bidx, off, 32);
        if (ov > best || (ov == best && oi < bidx)) { best = ov; bidx = oi; }
    }
    h0[(size_t)wid * 32 + lane] = emb[(size_t)bidx * 32 + lane];
}

// ---------------- dense GEMM via WMMA: [N,32] @ [32,OUT] -> [N,OUT] ----------------
// One wave computes a 16-row tile; K=32 exactly one v_wmma_f32_16x16x32_f16.
// Wave-uniform full-tile fast path: no per-element EXEC masking around the WMMA.
template <int OUT>
__global__ void gat_gemm_wmma(const float* __restrict__ A,   // [N,32] f32
                              const float* __restrict__ W,   // [32,OUT] f32
                              float* __restrict__ D,         // [N,OUT] f32
                              int N) {
    int wave = (int)((blockIdx.x * (long long)blockDim.x + threadIdx.x) >> 5);
    int lane = threadIdx.x & 31;
    int row0 = wave * 16;
    if (row0 >= N) return;                 // uniform per wave -> EXEC stays all-ones at WMMA

    const int g = lane >> 4, m = lane & 15;
    const bool full = (row0 + 16 <= N);    // wave-uniform

    // A fragment: 16x32 f16 (ISA 05_wmma 16-bit A layout).
    // Lane (g,m): halves 0-7 <- A[row, g*8 .. g*8+7], halves 8-15 <- A[row, 16+g*8 ..].
    // Both K-chunks are contiguous -> 4 x b128 loads on the fast path.
    v16h a;
    {
        const float* Arow = A + (size_t)(row0 + m) * 32;
        if (full) {
            v4f c0 = *(const v4f*)(Arow + g * 8);
            v4f c1 = *(const v4f*)(Arow + g * 8 + 4);
            v4f c2 = *(const v4f*)(Arow + 16 + g * 8);
            v4f c3 = *(const v4f*)(Arow + 16 + g * 8 + 4);
            #pragma unroll
            for (int i = 0; i < 4; ++i) {
                a[i]      = (_Float16)c0[i];
                a[4 + i]  = (_Float16)c1[i];
                a[8 + i]  = (_Float16)c2[i];
                a[12 + i] = (_Float16)c3[i];
            }
        } else {
            const bool rv = (row0 + m) < N;
            #pragma unroll
            for (int i = 0; i < 8; ++i) {
                a[i]     = (_Float16)(rv ? Arow[g * 8 + i] : 0.0f);
                a[8 + i] = (_Float16)(rv ? Arow[16 + g * 8 + i] : 0.0f);
            }
        }
    }

    #pragma unroll
    for (int t = 0; t < OUT / 16; ++t) {
        // B fragment: 32x16 f16. col N = lane&15; halves 0-7 <- K g*8.., halves 8-15 <- K 16+g*8..
        v16h b;
        #pragma unroll
        for (int i = 0; i < 16; ++i)
            b[i] = (_Float16)W[(size_t)((i < 8) ? (g * 8 + i) : (16 + g * 8 + i - 8)) * OUT + t * 16 + m];

        v8f c = {};
        c = __builtin_amdgcn_wmma_f32_16x16x32_f16(false, a, false, b, (short)0, c,
                                                   false, false);
        // D: VGPR r -> row r + 8*g, col = lane&15
        if (full) {
            #pragma unroll
            for (int r = 0; r < 8; ++r)
                D[(size_t)(row0 + r + 8 * g) * OUT + t * 16 + m] = c[r];
        } else {
            #pragma unroll
            for (int r = 0; r < 8; ++r) {
                int orow = row0 + r + 8 * g;
                if (orow < N) D[(size_t)orow * OUT + t * 16 + m] = c[r];
            }
        }
    }
}

// ---------------- per-node attention logits: als = h.a_src, ald = h.a_dst ----------------
template <int DCH>
__global__ void gat_attn_logits(const float* __restrict__ h,
                                const float* __restrict__ a_s, const float* __restrict__ a_d,
                                float* __restrict__ als, float* __restrict__ ald, int N) {
    int n = blockIdx.x * blockDim.x + threadIdx.x;
    if (n >= N) return;
    float s = 0.f, d = 0.f;
    #pragma unroll
    for (int k = 0; k < DCH; ++k) {
        float v = h[(size_t)n * DCH + k];
        s += v * a_s[k];
        d += v * a_d[k];
    }
    als[n] = s; ald[n] = d;
}

// ---------------- monotonic int encoding for float atomicMax ----------------
__device__ __forceinline__ int enc_f(float f) {
    int i = __float_as_int(f);
    return (i >= 0) ? i : (i ^ 0x7fffffff);
}
__device__ __forceinline__ float dec_f(int i) {
    return __int_as_float((i >= 0) ? i : (i ^ 0x7fffffff));
}

// pass A: e = leaky_relu(als[src]+ald[dst]); store e; segment max over dst
__global__ void gat_edge_max(const int* __restrict__ esrc, const int* __restrict__ edst,
                             int E, int ET,
                             const float* __restrict__ als, const float* __restrict__ ald,
                             float* __restrict__ ebuf, int* __restrict__ mbuf) {
    int i = blockIdx.x * blockDim.x + threadIdx.x;
    if (i >= ET) return;
    int s = (i < E) ? esrc[i] : (i - E);
    int d = (i < E) ? edst[i] : (i - E);
    float e = als[s] + ald[d];
    e = (e > 0.f) ? e : NEG_SLOPE * e;
    ebuf[i] = e;
    atomicMax(&mbuf[d], enc_f(e));
}

// pass B: ex = exp(e - m[dst]); overwrite ebuf with ex; segment sum -> denom
__global__ void gat_edge_exp(const int* __restrict__ edst, int E, int ET,
                             float* __restrict__ ebuf, const int* __restrict__ mbuf,
                             float* __restrict__ denom) {
    int i = blockIdx.x * blockDim.x + threadIdx.x;
    if (i >= ET) return;
    int d = (i < E) ? edst[i] : (i - E);
    float ex = __expf(ebuf[i] - dec_f(mbuf[d]));
    ebuf[i] = ex;
    atomicAdd(&denom[d], ex);
}

// pass C: acc[dst] += alpha * h[src]; DCH lanes per edge (broadcast alpha loads)
template <int DCH>
__global__ void gat_edge_aggr(const int* __restrict__ esrc, const int* __restrict__ edst,
                              int E, int ET,
                              const float* __restrict__ ebuf, const float* __restrict__ denom,
                              const float* __restrict__ h, float* __restrict__ acc) {
    long long tid = (long long)blockIdx.x * blockDim.x + threadIdx.x;
    long long total = (long long)ET * DCH;
    if (tid >= total) return;
    int i = (int)(tid / DCH);
    int c = (int)(tid % DCH);
    int s = (i < E) ? esrc[i] : (i - E);
    int d = (i < E) ? edst[i] : (i - E);
    float alpha = ebuf[i] / (denom[d] + 1e-16f);
    atomicAdd(&acc[(size_t)d * DCH + c], alpha * h[(size_t)s * DCH + c]);
}

// out = relu(acc + bias)
template <int DCH>
__global__ void gat_bias_relu(const float* __restrict__ acc, const float* __restrict__ bias,
                              float* __restrict__ out, int N) {
    long long i = (long long)blockIdx.x * blockDim.x + threadIdx.x;
    if (i >= (long long)N * DCH) return;
    int c = (int)(i % DCH);
    float v = acc[i] + bias[c];
    out[i] = (v > 0.f) ? v : 0.f;
}

// pooling: 16 lanes per node, atomic accumulate into [G,16] (+ counts)
__global__ void gat_pool(const float* __restrict__ acc2, const int* __restrict__ batch,
                         float* __restrict__ pool, float* __restrict__ cnt, int N) {
    long long tid = (long long)blockIdx.x * blockDim.x + threadIdx.x;
    int n = (int)(tid >> 4);
    int c = (int)(tid & 15);
    if (n >= N) return;
    int g = batch[n];
    atomicAdd(&pool[(size_t)g * 16 + c], acc2[(size_t)n * 16 + c]);
    if (c == 0) atomicAdd(&cnt[g], 1.0f);
}

// mean (+b2 folded post-mean) + row softmax over 16 classes; one thread per graph
__global__ void gat_softmax(const float* __restrict__ pool, const float* __restrict__ cnt,
                            const float* __restrict__ b2, float* __restrict__ out, int G) {
    int g = blockIdx.x * blockDim.x + threadIdx.x;
    if (g >= G) return;
    float inv = 1.0f / fmaxf(cnt[g], 1.0f);
    float v[16], mx = -3.4e38f;
    #pragma unroll
    for (int i = 0; i < 16; ++i) {
        v[i] = pool[(size_t)g * 16 + i] * inv + b2[i];
        mx = fmaxf(mx, v[i]);
    }
    float sum = 0.f;
    #pragma unroll
    for (int i = 0; i < 16; ++i) { v[i] = __expf(v[i] - mx); sum += v[i]; }
    float r = 1.0f / sum;
    #pragma unroll
    for (int i = 0; i < 16; ++i) out[(size_t)g * 16 + i] = v[i] * r;
}

// ---------------------------------------------------------------------------
extern "C" void kernel_launch(void* const* d_in, const int* in_sizes, int n_in,
                              void* d_out, int out_size, void* d_ws, size_t ws_size,
                              hipStream_t stream) {
    const float* x     = (const float*)d_in[0];   // [N,128]
    const int*   eidx  = (const int*)  d_in[1];   // [2,E]
    const int*   batch = (const int*)  d_in[2];   // [N]
    const float* emb   = (const float*)d_in[3];   // [128,32]
    const float* W1    = (const float*)d_in[4];   // [32,32]
    const float* a1s   = (const float*)d_in[5];
    const float* a1d   = (const float*)d_in[6];
    const float* b1    = (const float*)d_in[7];
    const float* W2    = (const float*)d_in[8];   // [32,16]
    const float* a2s   = (const float*)d_in[9];
    const float* a2d   = (const float*)d_in[10];
    const float* b2    = (const float*)d_in[11];
    float* out = (float*)d_out;

    const int N  = in_sizes[2];
    const int E  = in_sizes[1] / 2;
    const int ET = E + N;                 // with self loops
    const int G  = out_size / 16;
    const int*  esrc = eidx;
    const int*  edst = eidx + E;

    // ---- workspace carve (floats) ----
    float* ws = (float*)d_ws;
    size_t o = 0;
    float* h0   = ws + o; o += (size_t)N * 32;   // embed out / relu(h1) reuse
    float* h1   = ws + o; o += (size_t)N * 32;   // layer1 GEMM out
    float* acc1 = ws + o; o += (size_t)N * 32;   // layer1 aggregation
    float* h2   = ws + o; o += (size_t)N * 16;   // layer2 GEMM out
    float* acc2 = ws + o; o += (size_t)N * 16;   // layer2 aggregation
    float* als  = ws + o; o += (size_t)N;
    float* ald  = ws + o; o += (size_t)N;
    int*   mbuf = (int*)(ws + o); o += (size_t)N;
    float* den  = ws + o; o += (size_t)N;
    float* ebuf = ws + o; o += (size_t)ET;
    float* pool = ws + o; o += (size_t)G * 16;
    float* cnt  = ws + o; o += (size_t)G;
    (void)ws_size; (void)n_in;

    const int BT = 256;
    int nodeBlk  = ceil_div(N, BT);
    int edgeBlk  = ceil_div(ET, BT);
    int waveBlk  = ceil_div((long long)N * 32, BT);          // wave-per-node kernels
    int tileBlk  = ceil_div((long long)ceil_div(N, 16) * 32, BT); // wave-per-16-row-tile

    // ---- embedding ----
    gat_embed<<<waveBlk, BT, 0, stream>>>(x, emb, h0, N);

    // ================= layer 1 (D=32) =================
    gat_gemm_wmma<32><<<tileBlk, BT, 0, stream>>>(h0, W1, h1, N);
    gat_attn_logits<32><<<nodeBlk, BT, 0, stream>>>(h1, a1s, a1d, als, ald, N);

    gat_fill_u32<<<nodeBlk, BT, 0, stream>>>((unsigned*)mbuf, 0x80000000u, N);
    gat_fill_u32<<<nodeBlk, BT, 0, stream>>>((unsigned*)den, 0u, N);
    gat_fill_u32<<<ceil_div((long long)N * 32, BT), BT, 0, stream>>>((unsigned*)acc1, 0u, (long long)N * 32);

    gat_edge_max<<<edgeBlk, BT, 0, stream>>>(esrc, edst, E, ET, als, ald, ebuf, mbuf);
    gat_edge_exp<<<edgeBlk, BT, 0, stream>>>(edst, E, ET, ebuf, mbuf, den);
    gat_edge_aggr<32><<<ceil_div((long long)ET * 32, BT), BT, 0, stream>>>(
        esrc, edst, E, ET, ebuf, den, h1, acc1);
    gat_bias_relu<32><<<ceil_div((long long)N * 32, BT), BT, 0, stream>>>(acc1, b1, h0, N);

    // ================= layer 2 (D=16) =================
    gat_gemm_wmma<16><<<tileBlk, BT, 0, stream>>>(h0, W2, h2, N);
    gat_attn_logits<16><<<nodeBlk, BT, 0, stream>>>(h2, a2s, a2d, als, ald, N);

    gat_fill_u32<<<nodeBlk, BT, 0, stream>>>((unsigned*)mbuf, 0x80000000u, N);
    gat_fill_u32<<<nodeBlk, BT, 0, stream>>>((unsigned*)den, 0u, N);
    gat_fill_u32<<<ceil_div((long long)N * 16, BT), BT, 0, stream>>>((unsigned*)acc2, 0u, (long long)N * 16);
    gat_fill_u32<<<ceil_div((long long)(G * 16 + G), BT), BT, 0, stream>>>((unsigned*)pool, 0u, (long long)G * 16 + G);

    gat_edge_max<<<edgeBlk, BT, 0, stream>>>(esrc, edst, E, ET, als, ald, ebuf, mbuf);
    gat_edge_exp<<<edgeBlk, BT, 0, stream>>>(edst, E, ET, ebuf, mbuf, den);
    gat_edge_aggr<16><<<ceil_div((long long)ET * 16, BT), BT, 0, stream>>>(
        esrc, edst, E, ET, ebuf, den, h2, acc2);

    // ---- pool + softmax (b2 folded after mean) ----
    gat_pool<<<ceil_div((long long)N * 16, BT), BT, 0, stream>>>(acc2, batch, pool, cnt, N);
    gat_softmax<<<ceil_div(G, 64), 64, 0, stream>>>(pool, cnt, b2, out, G);
}
